// EdgeNet_21036749816032
// MI455X (gfx1250) — matrix operand order repaired and checked by reading, not compiled, not run on hardware
//
#include <hip/hip_runtime.h>
#include <hip/hip_bf16.h>

// ---------------- model constants ----------------
#define NN      8192      // N = B*NPG nodes
#define DD      128       // inner size
#define BB      128       // boards
#define NPG     64        // nodes per graph
#define EMAIN   1858      // action edges per graph
#define EOTH    512       // other edges per graph
#define NACT    4672
#define NLAYER  8
#define FIVED   640
#define EPS     1e-5f

#define NE_MAIN (BB*EMAIN)   // 237824
#define NE_OTH  (BB*EOTH)    // 65536

// ---------------- WMMA types ----------------
typedef __bf16 bf16_t;
typedef __attribute__((ext_vector_type(16))) __bf16 v16bf;
typedef __attribute__((ext_vector_type(8)))  __bf16 v8bf;
typedef __attribute__((ext_vector_type(8)))  float  v8f;

// LDS row stride in bf16 elements; 40*2 = 80 bytes, multiple of 16 -> b128-aligned rows
#define LSTR 40

// =====================================================================
// Generic GEMM: C[M x 128] = A[M x K] @ W[K x 128] + bias (+ resid)
// A row-major lda=K (fp32), W row-major [K][128] (fp32). Output fp32 ld=128.
// grid.x = M/32, block = 256 threads (8 waves). K must be multiple of 32.
// Each wave computes two 16x16 C tiles (rows m0..15 and m0+16..31) via
// v_wmma_f32_16x16x32_bf16, sharing one B fragment.
// =====================================================================
template<bool RES>
__global__ void k_gemm(const float* __restrict__ A, const float* __restrict__ W,
                       const float* __restrict__ bias, const float* __restrict__ resid,
                       float* __restrict__ C, int K)
{
    __shared__ __align__(16) bf16_t As[32 * LSTR];    // A tile 32x32 (row-major, padded)
    __shared__ __align__(16) bf16_t Ws[128 * LSTR];   // W tile stored transposed: [n][k]

    const int tid   = threadIdx.x;
    const int wave  = tid >> 5;
    const int lane  = tid & 31;
    const int l16   = lane & 15;          // A row (local) / B+C column (local)
    const int khalf = lane >> 4;          // 0/1 half selector
    const int m0    = blockIdx.x * 32;
    const int ncol  = wave * 16 + l16;    // global output column (0..127)

    v8f acc0, acc1;
    {
        float bv = bias[ncol];
        #pragma unroll
        for (int i = 0; i < 8; ++i) { acc0[i] = bv; acc1[i] = bv; }
    }

    for (int k0 = 0; k0 < K; k0 += 32) {
        // ---- stage A tile: 32 rows x 32 k = 512 float2-pairs, packed b32 stores
        #pragma unroll
        for (int i = 0; i < 2; ++i) {
            int p  = tid + i * 256;       // pair index
            int rr = p >> 4, cp = p & 15; // row, k-pair (lanes -> consecutive pairs: coalesced)
            float2 f = *reinterpret_cast<const float2*>(
                A + (size_t)(m0 + rr) * K + k0 + 2 * cp);
            union { bf16_t h[2]; unsigned u; } pk;
            pk.h[0] = (bf16_t)f.x; pk.h[1] = (bf16_t)f.y;
            *reinterpret_cast<unsigned*>(As + rr * LSTR + 2 * cp) = pk.u;
        }
        // ---- stage W tile transposed: Ws[n][k], 2048 k-pairs, packed b32 stores
        #pragma unroll
        for (int i = 0; i < 8; ++i) {
            int p  = tid + i * 256;
            int nn = p & 127, kp = p >> 7; // lanes -> consecutive n: coalesced global loads
            float fa = W[(size_t)(k0 + 2 * kp) * DD + nn];
            float fb = W[(size_t)(k0 + 2 * kp + 1) * DD + nn];
            union { bf16_t h[2]; unsigned u; } pk;
            pk.h[0] = (bf16_t)fa; pk.h[1] = (bf16_t)fb;
            *reinterpret_cast<unsigned*>(Ws + nn * LSTR + 2 * kp) = pk.u;
        }
        // prefetch next k-tile while we compute this one
        if (k0 + 32 < K) {
            __builtin_prefetch(A + (size_t)(m0 + (tid >> 4)) * K + k0 + 32, 0, 0);
            __builtin_prefetch(W + (size_t)(k0 + 32 + (tid >> 7)) * DD + (tid & 127), 0, 0);
        }
        __syncthreads();

        // A fragments (16-bit A 16x32 layout): lane holds row l16 (+16 for tile 1),
        // e=0..7 -> k = khalf*8 + e ; e=8..15 -> k = 16 + khalf*8 + (e-8)
        v8bf a0lo = *reinterpret_cast<const v8bf*>(As + l16 * LSTR + khalf * 8);
        v8bf a0hi = *reinterpret_cast<const v8bf*>(As + l16 * LSTR + 16 + khalf * 8);
        v8bf a1lo = *reinterpret_cast<const v8bf*>(As + (16 + l16) * LSTR + khalf * 8);
        v8bf a1hi = *reinterpret_cast<const v8bf*>(As + (16 + l16) * LSTR + 16 + khalf * 8);
        // B fragment (16-bit B 32x16 layout): lane holds column l16, same k order as A
        v8bf blo  = *reinterpret_cast<const v8bf*>(Ws + ncol * LSTR + khalf * 8);
        v8bf bhi  = *reinterpret_cast<const v8bf*>(Ws + ncol * LSTR + 16 + khalf * 8);

        v16bf a0, a1, b;
        #pragma unroll
        for (int i = 0; i < 8; ++i) {
            a0[i] = a0lo[i]; a0[i + 8] = a0hi[i];
            a1[i] = a1lo[i]; a1[i + 8] = a1hi[i];
            b[i]  = blo[i];  b[i + 8]  = bhi[i];
        }

        acc0 = __builtin_amdgcn_wmma_f32_16x16x32_bf16(
                   false, a0, false, b, (short)0, acc0, false, false);
        acc1 = __builtin_amdgcn_wmma_f32_16x16x32_bf16(
                   false, a1, false, b, (short)0, acc1, false, false);
        __syncthreads();
    }

    // C layout: acc[r] -> (m = m0 + khalf*8 + r, n = ncol); tile1 at +16 rows
    #pragma unroll
    for (int r = 0; r < 8; ++r) {
        int m = m0 + khalf * 8 + r;
        float v0 = acc0[r];
        float v1 = acc1[r];
        if (RES) {
            v0 += resid[(size_t)m * DD + ncol];
            v1 += resid[(size_t)(m + 16) * DD + ncol];
        }
        C[(size_t)m * DD + ncol] = v0;
        C[(size_t)(m + 16) * DD + ncol] = v1;
    }
}

// =====================================================================
// small elementwise / graph kernels
// =====================================================================
__global__ void k_zero(float* p, int n) {
    int t = blockIdx.x * blockDim.x + threadIdx.x;
    if (t < n) p[t] = 0.f;
}

__global__ void k_count(const int* __restrict__ s, const int* __restrict__ r,
                        float* degS, float* degR, int nE) {
    int t = blockIdx.x * blockDim.x + threadIdx.x;
    if (t >= nE) return;
    atomicAdd(degS + s[t], 1.f);
    atomicAdd(degR + r[t], 1.f);
}

__global__ void k_norm(float* deg, int n) {   // rsqrt(deg + 1): self edge included
    int t = blockIdx.x * blockDim.x + threadIdx.x;
    if (t < n) deg[t] = rsqrtf(deg[t] + 1.0f);
}

__global__ void k_embed(const int* __restrict__ ids, const float* __restrict__ emb,
                        float* __restrict__ nodes) {
    int t = blockIdx.x * blockDim.x + threadIdx.x;   // NN*DD
    nodes[t] = emb[(size_t)ids[t >> 7] * DD + (t & 127)];
}

// h block init with self-edge contribution: h[i][blk*128+d] = x[i][d] * snorm[i]
__global__ void k_self_init(const float* __restrict__ x, const float* __restrict__ sn,
                            float* __restrict__ hblk) {
    int t = blockIdx.x * blockDim.x + threadIdx.x;   // NN*DD
    int i = t >> 7, d = t & 127;
    hblk[(size_t)i * FIVED + d] = x[t] * sn[i];
}

// edge scatter-add: h[r][d] += x[s][d]*snorm[s]; one wave per edge, 4 floats/lane
__global__ void k_scatter(const float* __restrict__ x, const int* __restrict__ se,
                          const int* __restrict__ re, const float* __restrict__ sn,
                          float* __restrict__ hblk, int nE) {
    int t = blockIdx.x * blockDim.x + threadIdx.x;
    int e = t >> 5, lane = t & 31;
    if (e >= nE) return;
    int s = se[e], r = re[e];
    float sc = sn[s];
    float4 xv = *reinterpret_cast<const float4*>(x + (size_t)s * DD + lane * 4);
    float* dst = hblk + (size_t)r * FIVED + lane * 4;
    atomicAdd(dst + 0, xv.x * sc);
    atomicAdd(dst + 1, xv.y * sc);
    atomicAdd(dst + 2, xv.z * sc);
    atomicAdd(dst + 3, xv.w * sc);
}

// in-place: h = relu(bn( h * rnorm[block] ))  over [NN, 640]
__global__ void k_bnrelu_h(float* __restrict__ h, const float* __restrict__ rn,
                           const float* __restrict__ bb, const float* __restrict__ bm,
                           const float* __restrict__ bs, const float* __restrict__ bv) {
    int t = blockIdx.x * blockDim.x + threadIdx.x;   // NN*FIVED
    int i = t / FIVED, c = t % FIVED;
    int rel = c >> 7;
    float v = h[t] * rn[(size_t)rel * NN + i];
    v = bs[c] * (v - bm[c]) * rsqrtf(bv[c] + EPS) + bb[c];
    h[t] = fmaxf(v, 0.f);
}

// out = relu(bn(in)) over [NN, 128]
__global__ void k_bnrelu(const float* __restrict__ in, float* __restrict__ out,
                         const float* __restrict__ bb, const float* __restrict__ bm,
                         const float* __restrict__ bs, const float* __restrict__ bv) {
    int t = blockIdx.x * blockDim.x + threadIdx.x;   // NN*DD
    int c = t & 127;
    float v = bs[c] * (in[t] - bm[c]) * rsqrtf(bv[c] + EPS) + bb[c];
    out[t] = fmaxf(v, 0.f);
}

// attention logit: attl[i] = dot(v[i,:], aw) + ab  (one wave per node)
__global__ void k_att(const float* __restrict__ v, const float* __restrict__ aw,
                      const float* __restrict__ ab, float* __restrict__ attl) {
    int t = blockIdx.x * blockDim.x + threadIdx.x;
    int node = t >> 5, lane = t & 31;
    float4 vv = *reinterpret_cast<const float4*>(v + (size_t)node * DD + lane * 4);
    float4 ww = *reinterpret_cast<const float4*>(aw + lane * 4);
    float s = vv.x * ww.x + vv.y * ww.y + vv.z * ww.z + vv.w * ww.w;
    #pragma unroll
    for (int o = 16; o > 0; o >>= 1) s += __shfl_down(s, o, 32);
    if (lane == 0) attl[node] = s + ab[0];
}

// edge logits + per-block partial min (8 waves/block, one edge per wave)
__global__ void k_elogit(const float* __restrict__ u, const float* __restrict__ w,
                         const float* __restrict__ ee, const int* __restrict__ se,
                         const int* __restrict__ re, const int* __restrict__ et,
                         float* __restrict__ elog, float* __restrict__ pmin) {
    __shared__ float smin[8];
    int wave = threadIdx.x >> 5, lane = threadIdx.x & 31;
    int e = blockIdx.x * 8 + wave;                   // grid sized exactly
    int s = se[e], r = re[e], c = et[e];
    float4 uu = *reinterpret_cast<const float4*>(u + (size_t)s * DD + lane * 4);
    float4 wv = *reinterpret_cast<const float4*>(w + (size_t)r * DD + lane * 4);
    float4 em = *reinterpret_cast<const float4*>(ee + (size_t)c * DD + lane * 4);
    float acc = uu.x * wv.x * em.x + uu.y * wv.y * em.y +
                uu.z * wv.z * em.z + uu.w * wv.w * em.w;
    #pragma unroll
    for (int o = 16; o > 0; o >>= 1) acc += __shfl_down(acc, o, 32);
    if (lane == 0) { elog[e] = acc; smin[wave] = acc; }
    __syncthreads();
    if (threadIdx.x == 0) {
        float m = smin[0];
        #pragma unroll
        for (int i = 1; i < 8; ++i) m = fminf(m, smin[i]);
        pmin[blockIdx.x] = m;
    }
}

__global__ void k_minreduce(const float* __restrict__ p, int n, float* __restrict__ out) {
    __shared__ float s[256];
    float m = 1e30f;
    for (int i = threadIdx.x; i < n; i += 256) m = fminf(m, p[i]);
    s[threadIdx.x] = m; __syncthreads();
    for (int st = 128; st > 0; st >>= 1) {
        if (threadIdx.x < st) s[threadIdx.x] = fminf(s[threadIdx.x], s[threadIdx.x + st]);
        __syncthreads();
    }
    if (threadIdx.x == 0) out[0] = s[0];
}

__global__ void k_fill(float* __restrict__ logits, const float* __restrict__ mv) {
    int t = blockIdx.x * blockDim.x + threadIdx.x;   // BB*NACT
    logits[t] = mv[0];
}

__global__ void k_scatter_logits(const float* __restrict__ elog, const int* __restrict__ ea,
                                 float* __restrict__ logits) {
    int t = blockIdx.x * blockDim.x + threadIdx.x;
    if (t >= NE_MAIN) return;
    int b = t / EMAIN;
    logits[(size_t)b * NACT + ea[t]] = elog[t];
}

// value head: one block (128 threads) per board; masked segment softmax pooling
__global__ void k_value(const float* __restrict__ v, const float* __restrict__ attl,
                        const float* __restrict__ vow, const float* __restrict__ vob,
                        float* __restrict__ out) {
    __shared__ float s_att[NPG];
    __shared__ float s_red[128];
    int b = blockIdx.x, t = threadIdx.x;
    int base = b * NPG;

    float al = -1e30f;
    if (t < NPG) {
        float x = attl[base + t];
        s_att[t] = x;
        al = (t == 0) ? -1e30f : x;     // node (i % 64 == 0) is masked
    }
    s_red[t] = al;
    __syncthreads();
    for (int st = 64; st > 0; st >>= 1) {
        if (t < st) s_red[t] = fmaxf(s_red[t], s_red[t + st]);
        __syncthreads();
    }
    float mx = s_red[0];
    __syncthreads();

    float e = 0.f;
    if (t < NPG && t != 0) e = expf(s_att[t] - mx);
    s_red[t] = (t < NPG) ? e : 0.f;
    __syncthreads();
    for (int st = 64; st > 0; st >>= 1) {
        if (t < st) s_red[t] += s_red[t + st];
        __syncthreads();
    }
    float denom = s_red[0];
    __syncthreads();
    if (t < NPG) s_att[t] = e / denom;
    __syncthreads();

    // pooled[d=t] = sum_j att[j] * v[base+j][t]; relu; dot with v_out.w; tanh
    float p = 0.f;
    for (int j = 0; j < NPG; ++j) p += v[(size_t)(base + j) * DD + t] * s_att[j];
    p = fmaxf(p, 0.f);
    s_red[t] = p * vow[t];
    __syncthreads();
    for (int st = 64; st > 0; st >>= 1) {
        if (t < st) s_red[t] += s_red[t + st];
        __syncthreads();
    }
    if (t == 0) out[b] = tanhf(s_red[0] + vob[0]);
}

// =====================================================================
// host-side orchestration
// =====================================================================
extern "C" void kernel_launch(void* const* d_in, const int* in_sizes, int n_in,
                              void* d_out, int out_size, void* d_ws, size_t ws_size,
                              hipStream_t stream)
{
    (void)in_sizes; (void)n_in; (void)out_size; (void)ws_size;
    const int* node_ids = (const int*)d_in[0];
    const int* senders  = (const int*)d_in[1];
    const int* receivs  = (const int*)d_in[2];
    const int* grid_s   = (const int*)d_in[3];
    const int* grid_r   = (const int*)d_in[4];
    const int* act_s    = (const int*)d_in[5];
    const int* act_r    = (const int*)d_in[6];
    const int* pas_s    = (const int*)d_in[7];
    const int* pas_r    = (const int*)d_in[8];
    const int* edges    = (const int*)d_in[9];
    const int* eact     = (const int*)d_in[10];
    // d_in[11] = n_node (all 64, hardcoded)

    // params pytree leaves (sorted dict keys, JAX flatten order), starting at 12
    #define PF(i) ((const float*)d_in[i])
    int pi = 12;
    const float* attd_b = PF(pi++);  const float* attd_w = PF(pi++);   // att_dense
    const float* dotu_b = PF(pi++);  const float* dotu_w = PF(pi++);
    const float* dotv_b = PF(pi++);  const float* dotv_w = PF(pi++);
    const float* eemb   = PF(pi++);                                     // edge_embed [4,128]
    const float* emb    = PF(pi++);                                     // embed [13,128]
    const float* hbn_b = PF(pi++); const float* hbn_m = PF(pi++);
    const float* hbn_s = PF(pi++); const float* hbn_v = PF(pi++);       // head_bn
    const float *Lbn_b[NLAYER], *Lbn_m[NLAYER], *Lbn_s[NLAYER], *Lbn_v[NLAYER];
    const float *Lgc_b[NLAYER][5], *Lgc_w[NLAYER][5];
    const float *Lout_b[NLAYER], *Lout_w[NLAYER];
    for (int l = 0; l < NLAYER; ++l) {
        Lbn_b[l] = PF(pi++); Lbn_m[l] = PF(pi++); Lbn_s[l] = PF(pi++); Lbn_v[l] = PF(pi++);
        for (int g = 0; g < 5; ++g) { Lgc_b[l][g] = PF(pi++); Lgc_w[l][g] = PF(pi++); }
        Lout_b[l] = PF(pi++); Lout_w[l] = PF(pi++);
    }
    const float* nbn_b = PF(pi++); const float* nbn_m = PF(pi++);
    const float* nbn_s = PF(pi++); const float* nbn_v = PF(pi++);       // nl_bn
    const float* nld_b = PF(pi++); const float* nld_w = PF(pi++);       // nl_dense
    const float* vbn_b = PF(pi++); const float* vbn_m = PF(pi++);
    const float* vbn_s = PF(pi++); const float* vbn_v = PF(pi++);       // v_bn
    const float* vd_b  = PF(pi++); const float* vd_w  = PF(pi++);       // v_dense
    const float* vo_b  = PF(pi++); const float* vo_w  = PF(pi++);       // v_out
    #undef PF

    // ---- workspace carve-up (floats) ----
    float* ws = (float*)d_ws;
    const size_t ND = (size_t)NN * DD;          // 1,048,576
    float* nodes = ws;               size_t o = ND;
    float* h     = ws + o;           o += (size_t)NN * FIVED;   // 5,242,880
    float* xtmp  = ws + o;           o += ND;
    float* snorm = ws + o;           o += 5 * (size_t)NN;       // degS -> snorm (in place)
    float* rnorm = ws + o;           o += 5 * (size_t)NN;       // degR -> rnorm
    float* attl  = ws + o;           o += NN;
    float* elog  = ws + o;           o += (size_t)NE_MAIN;
    float* pmin  = ws + o;           o += 32768;
    float* minv  = ws + o;           o += 16;
    // head buffers reuse h (5*ND == NN*FIVED exactly)
    float* xhead = h;
    float* nlb   = h + ND;
    float* ub    = h + 2 * ND;
    float* wb    = h + 3 * ND;
    float* vb    = h + 4 * ND;

    float* logits = (float*)d_out;              // [BB, NACT]
    float* val    = logits + (size_t)BB * NACT; // [BB]

    const int* relS[5] = {senders, receivs, grid_s, act_s, pas_s};
    const int* relR[5] = {receivs, senders, grid_r, act_r, pas_r};
    const int  relE[5] = {NE_MAIN, NE_MAIN, NE_OTH, NE_OTH, NE_OTH};

    const int T = 256;

    // ---- degree norms (constant across layers) ----
    k_zero<<<(10 * NN) / T, T, 0, stream>>>(snorm, 10 * NN);
    for (int rel = 0; rel < 5; ++rel)
        k_count<<<(relE[rel] + T - 1) / T, T, 0, stream>>>(
            relS[rel], relR[rel], snorm + (size_t)rel * NN, rnorm + (size_t)rel * NN, relE[rel]);
    k_norm<<<(10 * NN) / T, T, 0, stream>>>(snorm, 10 * NN);

    // ---- embedding ----
    k_embed<<<ND / T, T, 0, stream>>>(node_ids, emb, nodes);

    // ---- GCN layers ----
    for (int l = 0; l < NLAYER; ++l) {
        for (int rel = 0; rel < 5; ++rel) {
            k_gemm<false><<<NN / 32, T, 0, stream>>>(nodes, Lgc_w[l][rel], Lgc_b[l][rel],
                                                     nullptr, xtmp, DD);
            float* hblk = h + (size_t)rel * DD;  // column block in [NN, 640]
            k_self_init<<<ND / T, T, 0, stream>>>(xtmp, snorm + (size_t)rel * NN, hblk);
            k_scatter<<<((size_t)relE[rel] * 32) / T, T, 0, stream>>>(
                xtmp, relS[rel], relR[rel], snorm + (size_t)rel * NN, hblk, relE[rel]);
        }
        k_bnrelu_h<<<((size_t)NN * FIVED) / T, T, 0, stream>>>(
            h, rnorm, Lbn_b[l], Lbn_m[l], Lbn_s[l], Lbn_v[l]);
        // nodes = h @ Wout + b + nodes (residual, in place)
        k_gemm<true><<<NN / 32, T, 0, stream>>>(h, Lout_w[l], Lout_b[l], nodes, nodes, FIVED);
    }

    // ---- heads ----
    k_bnrelu<<<ND / T, T, 0, stream>>>(nodes, xhead, hbn_b, hbn_m, hbn_s, hbn_v);

    k_gemm<false><<<NN / 32, T, 0, stream>>>(xhead, nld_w, nld_b, nullptr, xtmp, DD);
    k_bnrelu<<<ND / T, T, 0, stream>>>(xtmp, nlb, nbn_b, nbn_m, nbn_s, nbn_v);
    k_gemm<false><<<NN / 32, T, 0, stream>>>(nlb, dotu_w, dotu_b, nullptr, ub, DD);
    k_gemm<false><<<NN / 32, T, 0, stream>>>(nlb, dotv_w, dotv_b, nullptr, wb, DD);

    k_gemm<false><<<NN / 32, T, 0, stream>>>(xhead, vd_w, vd_b, nullptr, xtmp, DD);
    k_bnrelu<<<ND / T, T, 0, stream>>>(xtmp, vb, vbn_b, vbn_m, vbn_s, vbn_v);

    // edge logits + global min + scatter into logits
    k_elogit<<<NE_MAIN / 8, T, 0, stream>>>(ub, wb, eemb, senders, receivs, edges, elog, pmin);
    k_minreduce<<<1, T, 0, stream>>>(pmin, NE_MAIN / 8, minv);
    k_fill<<<((size_t)BB * NACT) / T, T, 0, stream>>>(logits, minv);
    k_scatter_logits<<<(NE_MAIN + T - 1) / T, T, 0, stream>>>(elog, eact, logits);

    // value head
    k_att<<<(NN * 32) / T, T, 0, stream>>>(vb, attd_w, attd_b, attl);
    k_value<<<BB, 128, 0, stream>>>(vb, attl, vo_w, vo_b, val);
}
